// UNCRTAINTS_Mamba3D_86088324481492
// MI455X (gfx1250) — matrix operand order repaired and checked by reading, not compiled, not run on hardware
//
#include <hip/hip_runtime.h>
#include <hip/hip_bf16.h>
#include <hip/hip_fp16.h>
#include <cstdint>
#include <cstddef>

// ---------------------------------------------------------------------------
// CDNA5 (gfx1250) implementation of the Mamba3D UNCRTAINTS forward pass.
// All matmul-shaped work (projections, im2col convs, 1x1 convs) runs through
// a single wave32 WMMA GEMM (v_wmma_f32_16x16x32_f16, f32 accumulate) whose
// LDS staging is double-buffered over the CDNA5 async-to-LDS path
// (GLOBAL_LOAD_ASYNC_TO_LDS_B128, ASYNCcnt / s_wait_asynccnt): the async DMA
// for tile k+1 overlaps the WMMAs of tile k.  GEMM operands are kept as f16
// in global memory (im2col emits f16, weights converted once, activation
// transposes folded into the f32->f16 conversion).
// ---------------------------------------------------------------------------

typedef __attribute__((ext_vector_type(16))) _Float16 v16h;
typedef __attribute__((ext_vector_type(8)))  float    v8f;
typedef __attribute__((ext_vector_type(2)))  _Float16 v2h;

enum { ACT_NONE = 0, ACT_RELU = 1, ACT_GELU = 2, ACT_SILU = 3, ACT_SOFTPLUS = 4 };

__device__ __forceinline__ float actf(float v, int act) {
  switch (act) {
    case ACT_RELU:     return fmaxf(v, 0.f);
    case ACT_GELU:     return 0.5f * v * (1.f + erff(v * 0.70710678118f));
    case ACT_SILU:     return v / (1.f + __expf(-v));
    case ACT_SOFTPLUS: return (v > 20.f) ? v : log1pf(__expf(v));
    default:           return v;
  }
}

// ---------------------------------------------------------------------------
// CDNA5 async-to-LDS helpers (ISA 15.18, tracked by ASYNCcnt)
// ---------------------------------------------------------------------------
__device__ __forceinline__ unsigned lds_off(const void* p) {
  // generic pointer to LDS: low 32 bits carry the LDS byte offset
  return (unsigned)(unsigned long long)p;
}
__device__ __forceinline__ void async_copy_b128(unsigned dst_lds, const void* src) {
  asm volatile("global_load_async_to_lds_b128 %0, %1, off"
               :: "v"(dst_lds), "v"(src) : "memory");
}
__device__ __forceinline__ void async_wait0() {
  asm volatile("s_wait_asynccnt 0x0" ::: "memory");
}

// ---------------------------------------------------------------------------
// WMMA fragment loaders (CDNA5 ISA 7.12.2 layouts, wave32)
// A (16x32 f16): lane m=lane&15; VGPR v<4 -> K = half*8+2v ; v>=4 -> 16+half*8+2(v-4)
// B (32x16 f16): lane n=lane&15; VGPR v<4 -> K = half*8+2v ; v>=4 -> 16+half*8+2(v-4)
// ---------------------------------------------------------------------------
#define LDK 40
#define BM 128
#define BN 64

__device__ __forceinline__ v16h load_a_frag(const _Float16* Asb, int m0, int lane) {
  const _Float16* rowp = Asb + (m0 + (lane & 15)) * LDK;
  const int half = (lane >> 4) * 8;
  v16h f;
#pragma unroll
  for (int v = 0; v < 8; ++v) {
    int k = (v < 4) ? (half + 2 * v) : (16 + half + 2 * (v - 4));
    v2h p = *(const v2h*)(rowp + k);
    f[2 * v] = p[0];
    f[2 * v + 1] = p[1];
  }
  return f;
}

__device__ __forceinline__ v16h load_b_frag(const _Float16* Bsb, int n0, int lane) {
  const _Float16* rowp = Bsb + (n0 + (lane & 15)) * LDK;
  const int kb = (lane >> 4) * 8;
  v16h f;
#pragma unroll
  for (int v = 0; v < 8; ++v) {
    v2h p = *(const v2h*)(rowp + kb + ((v < 4) ? 2 * v : 16 + 2 * (v - 4)));
    f[2 * v] = p[0];
    f[2 * v + 1] = p[1];
  }
  return f;
}

// ---------------------------------------------------------------------------
// GEMM:  C(f32) = act( A(f16, MxK row-major) * B(f16, NxK row-major)^T + bias )
// Block = 128x64 tile, 256 threads = 8 waves (4 M-strips x 2 N-strips);
// each wave owns 32x32 -> four v_wmma_f32_16x16x32_f16 per K step with 2x
// fragment reuse.  Double-buffered staging: the async-to-LDS copy of tile
// k+1 is issued right after the barrier and overlaps the WMMAs of tile k;
// one barrier per K step.  Ragged edges use guarded scalar f16 loads.
// ---------------------------------------------------------------------------
__global__ __launch_bounds__(256) void k_gemm_wmma(
    const _Float16* __restrict__ A, const _Float16* __restrict__ B,
    float* __restrict__ C, const float* __restrict__ bias,
    int M, int N, int K, int lda, int ldb, int ldc, int tc, int act)
{
  __shared__ alignas(16) _Float16 As[2][BM * LDK];
  __shared__ alignas(16) _Float16 Bs[2][BN * LDK];

  const int tid  = threadIdx.x;
  const int lane = tid & 31;
  const int wave = tid >> 5;
  const int wm   = (wave & 3) * 32;   // wave's 32-row strip
  const int wn   = (wave >> 2) * 32;  // wave's 32-col strip
  const int m0   = blockIdx.y * BM;
  const int n0   = blockIdx.x * BN;

  const bool fastA = ((lda & 7) == 0) && (m0 + BM <= M);
  const bool fastB = ((ldb & 7) == 0) && (n0 + BN <= N);

  auto stageA = [&](int buf, int k0) {
    _Float16* dst = As[buf];
    if (fastA && (k0 + 32 <= K)) {
#pragma unroll
      for (int j = 0; j < 2; ++j) {
        int i = tid + j * 256;        // 512 16B chunks
        int r = i >> 2;               // row 0..127
        int q = (i & 3) * 8;          // k offset 0,8,16,24
        async_copy_b128(lds_off(&dst[r * LDK + q]),
                        A + (size_t)(m0 + r) * lda + k0 + q);
      }
    } else {
      for (int i = tid; i < BM * 32; i += 256) {
        int r = i >> 5, k = i & 31;
        int gm = m0 + r, gk = k0 + k;
        _Float16 v = (_Float16)0.f;
        if (gm < M && gk < K) v = A[(size_t)gm * lda + gk];
        dst[r * LDK + k] = v;
      }
    }
  };

  auto stageB = [&](int buf, int k0) {
    _Float16* dst = Bs[buf];
    if (fastB && (k0 + 32 <= K)) {
      int r = tid >> 2;               // n 0..63
      int q = (tid & 3) * 8;          // k offset
      async_copy_b128(lds_off(&dst[r * LDK + q]),
                      B + (size_t)(n0 + r) * ldb + k0 + q);
    } else {
      for (int i = tid; i < BN * 32; i += 256) {
        int r = i >> 5, k = i & 31;
        int gc = n0 + r, gk = k0 + k;
        _Float16 v = (_Float16)0.f;
        if (gc < N && gk < K) v = B[(size_t)gc * ldb + gk];
        dst[r * LDK + k] = v;
      }
    }
  };

  v8f acc[2][2];
#pragma unroll
  for (int a = 0; a < 2; ++a)
#pragma unroll
    for (int b = 0; b < 2; ++b)
      acc[a][b] = (v8f){0.f, 0.f, 0.f, 0.f, 0.f, 0.f, 0.f, 0.f};

  const int nk = (K + 31) / 32;
  // prologue: stage tile 0 into buffer 0
  stageA(0, 0);
  stageB(0, 0);

  for (int it = 0; it < nk; ++it) {
    const int k0  = it * 32;
    const int cur = it & 1;

    async_wait0();      // this wave's async writes for tile `it` landed
    __syncthreads();    // all waves staged tile `it`; prior reads retired

    // issue the async staging of tile it+1 into the other buffer;
    // it proceeds concurrently with the WMMAs below.
    if (it + 1 < nk) {
      stageA(cur ^ 1, k0 + 32);
      stageB(cur ^ 1, k0 + 32);
    }
    // speculative prefetch two tiles ahead (global_prefetch_b8)
    if (it + 2 < nk) {
      int r = tid & (BM - 1);
      if (m0 + r < M)
        __builtin_prefetch((const void*)&A[(size_t)(m0 + r) * lda + (k0 + 64)], 0, 1);
    }

    v16h a0 = load_a_frag(As[cur], wm, lane);
    v16h a1 = load_a_frag(As[cur], wm + 16, lane);
    v16h b0 = load_b_frag(Bs[cur], wn, lane);
    v16h b1 = load_b_frag(Bs[cur], wn + 16, lane);
    acc[0][0] = __builtin_amdgcn_wmma_f32_16x16x32_f16(false, a0, false, b0, (short)0, acc[0][0], false, false);
    acc[0][1] = __builtin_amdgcn_wmma_f32_16x16x32_f16(false, a0, false, b1, (short)0, acc[0][1], false, false);
    acc[1][0] = __builtin_amdgcn_wmma_f32_16x16x32_f16(false, a1, false, b0, (short)0, acc[1][0], false, false);
    acc[1][1] = __builtin_amdgcn_wmma_f32_16x16x32_f16(false, a1, false, b1, (short)0, acc[1][1], false, false);
  }

  // ---- epilogue: C/D 16x16 f32 layout: VGPR v holds M = v + 8*(lane>>4) ----
  const int half = lane >> 4;
  const int rN   = lane & 15;
#pragma unroll
  for (int mi = 0; mi < 2; ++mi)
#pragma unroll
    for (int ni = 0; ni < 2; ++ni)
#pragma unroll
      for (int v = 0; v < 8; ++v) {
        int r = m0 + wm + mi * 16 + half * 8 + v;
        int c = n0 + wn + ni * 16 + rN;
        if (r < M && c < N) {
          float val = acc[mi][ni][v] + (bias ? bias[c] : 0.f);
          val = actf(val, act);
          if (tc) C[(size_t)c * ldc + r] = val;
          else    C[(size_t)r * ldc + c] = val;
        }
      }
}

// ---------------------------------------------------------------------------
// conversion / transpose kernels (feed the f16 GEMM)
// ---------------------------------------------------------------------------
__global__ void k_cvt_h(const float* __restrict__ in, _Float16* __restrict__ out, long long n)
{
  long long i = (long long)blockIdx.x * blockDim.x + threadIdx.x;
  if (i < n) out[i] = (_Float16)in[i];
}

// (C,S) f32 channel-major -> (S,C) f16 row-major
__global__ void k_cvt_tr_h(const float* __restrict__ in, _Float16* __restrict__ out,
                           int C, long long S)
{
  long long i = (long long)blockIdx.x * blockDim.x + threadIdx.x;
  if (i >= (long long)C * S) return;
  long long l = i / C; int c = (int)(i % C);
  out[i] = (_Float16)in[(long long)c * S + l];
}

// reverse leading dim of (L,C) f16
__global__ void k_fliprows_h(const _Float16* __restrict__ in, _Float16* __restrict__ out,
                             int L, int C)
{
  long long i = (long long)blockIdx.x * blockDim.x + threadIdx.x;
  if (i >= (long long)L * C) return;
  long long l = i / C, c = i % C;
  out[i] = in[(long long)(L - 1 - l) * C + c];
}

// ---------------------------------------------------------------------------
// im2col for 3D (and 2D: KD=1) conv, SAME padding; emits f16 patches.
// dst[M = D*H*W][K = Cin*KD*KH*KW]; src element (c,d,h,w) at c*sC + d*sD + h*W + w
// ---------------------------------------------------------------------------
__global__ void k_im2col(const float* __restrict__ src, _Float16* __restrict__ dst,
                         int Cin, int D, int H, int W, int KD, int KH, int KW,
                         long long sC, long long sD)
{
  long long Kc = (long long)Cin * KD * KH * KW;
  long long M  = (long long)D * H * W;
  long long idx = (long long)blockIdx.x * blockDim.x + threadIdx.x;
  if (idx >= M * Kc) return;
  long long m = idx / Kc;
  int kk = (int)(idx % Kc);
  int w = (int)(m % W);
  int h = (int)((m / W) % H);
  int d = (int)(m / ((long long)W * H));
  int kw = kk % KW;
  int kh = (kk / KW) % KH;
  int kd = (kk / (KW * KH)) % KD;
  int c  = kk / (KW * KH * KD);
  int dd = d + kd - KD / 2, hh = h + kh - KH / 2, ww = w + kw - KW / 2;
  float v = 0.f;
  if (dd >= 0 && dd < D && hh >= 0 && hh < H && ww >= 0 && ww < W)
    v = src[(long long)c * sC + (long long)dd * sD + (long long)hh * W + ww];
  dst[m * Kc + kk] = (_Float16)v;
}

// ---------------------------------------------------------------------------
// GroupNorm (one block per group) with fused activation, layout (C, Sp)
// ---------------------------------------------------------------------------
__global__ __launch_bounds__(256) void k_groupnorm(
    const float* __restrict__ x, float* __restrict__ out,
    const float* __restrict__ w, const float* __restrict__ b,
    int C, long long Sp, int groups, int act)
{
  __shared__ float ssum[256], ssq[256];
  int g = blockIdx.x;
  int cpg = C / groups;
  long long n = (long long)cpg * Sp;
  const float* xg = x + (long long)g * n;
  float s = 0.f, q = 0.f;
  for (long long i = threadIdx.x; i < n; i += 256) { float v = xg[i]; s += v; q += v * v; }
  ssum[threadIdx.x] = s; ssq[threadIdx.x] = q;
  __syncthreads();
  for (int o = 128; o > 0; o >>= 1) {
    if ((int)threadIdx.x < o) { ssum[threadIdx.x] += ssum[threadIdx.x + o]; ssq[threadIdx.x] += ssq[threadIdx.x + o]; }
    __syncthreads();
  }
  float mean = ssum[0] / (float)n;
  float var  = ssq[0] / (float)n - mean * mean;
  float inv  = rsqrtf(var + 1e-5f);
  for (long long i = threadIdx.x; i < n; i += 256) {
    int c = g * cpg + (int)(i / Sp);
    float v = (xg[i] - mean) * inv * w[c] + b[c];
    out[(long long)g * n + i] = actf(v, act);
  }
}

// ---------------------------------------------------------------------------
// Mamba pieces
// ---------------------------------------------------------------------------
// causal depthwise conv1d (D_CONV=4) + SiLU; xi = columns [0,256) of xz (L,512)
// writes f32 (for the scan) and f16 (for the W_x GEMM)
__global__ void k_conv1d_silu(const float* __restrict__ xz, const float* __restrict__ w,
                              const float* __restrict__ b, float* __restrict__ xc,
                              _Float16* __restrict__ xch, int L)
{
  long long i = (long long)blockIdx.x * blockDim.x + threadIdx.x;
  if (i >= (long long)L * 256) return;
  int t = (int)(i >> 8), d = (int)(i & 255);
  float acc = b[d];
#pragma unroll
  for (int j = 0; j < 4; ++j) {
    int tt = t - 3 + j;
    if (tt >= 0) acc += xz[(long long)tt * 512 + d] * w[d * 4 + j];
  }
  float r = acc / (1.f + __expf(-acc));
  xc[i]  = r;
  xch[i] = (_Float16)r;
}

// selective scan: one lane per channel d, 16 states in registers
__global__ void k_scan(const float* __restrict__ dt, const float* __restrict__ xdb,
                       const float* __restrict__ xc, const float* __restrict__ A_log,
                       const float* __restrict__ Dp, float* __restrict__ ys, int L)
{
  int d = blockIdx.x * blockDim.x + threadIdx.x;
  if (d >= 256) return;
  float A[16], h[16];
#pragma unroll
  for (int n = 0; n < 16; ++n) { A[n] = -__expf(A_log[d * 16 + n]); h[n] = 0.f; }
  float Dv = Dp[d];
  for (int t = 0; t < L; ++t) {
    float dtv = dt[(long long)t * 256 + d];
    float u   = xc[(long long)t * 256 + d];
    const float* Bp = xdb + (long long)t * 40 + 8;
    const float* Cp = xdb + (long long)t * 40 + 24;
    float du = dtv * u, y = 0.f;
#pragma unroll
    for (int n = 0; n < 16; ++n) {
      h[n] = h[n] * __expf(dtv * A[n]) + du * Bp[n];
      y += h[n] * Cp[n];
    }
    ys[(long long)t * 256 + d] = y + u * Dv;
  }
}

// y(f16) = ys * silu(z); z = columns [256,512) of xz
__global__ void k_gate_h(const float* __restrict__ ys, const float* __restrict__ xz,
                         _Float16* __restrict__ y, int L)
{
  long long i = (long long)blockIdx.x * blockDim.x + threadIdx.x;
  if (i >= (long long)L * 256) return;
  int t = (int)(i >> 8), d = (int)(i & 255);
  float z = xz[(long long)t * 512 + 256 + d];
  y[i] = (_Float16)(ys[i] * (z / (1.f + __expf(-z))));
}

// ---------------------------------------------------------------------------
// elementwise / shape kernels (f32)
// ---------------------------------------------------------------------------
__global__ void k_axpby(const float* __restrict__ x, const float* __restrict__ y,
                        float* __restrict__ out, float a, float b, long long n)
{
  long long i = (long long)blockIdx.x * blockDim.x + threadIdx.x;
  if (i < n) out[i] = a * x[i] + b * y[i];
}

__global__ void k_add3(const float* __restrict__ x, const float* __restrict__ y,
                       const float* __restrict__ z, float* __restrict__ out, long long n)
{
  long long i = (long long)blockIdx.x * blockDim.x + threadIdx.x;
  if (i < n) out[i] = x[i] + y[i] + z[i];
}

__global__ void k_add_act(const float* __restrict__ x, const float* __restrict__ y,
                          float* __restrict__ out, int act, long long n)
{
  long long i = (long long)blockIdx.x * blockDim.x + threadIdx.x;
  if (i < n) out[i] = actf(x[i] + y[i], act);
}

__global__ void k_flipcols(const float* __restrict__ x, float* __restrict__ out,
                           int R, long long Cc)
{
  long long i = (long long)blockIdx.x * blockDim.x + threadIdx.x;
  if (i >= (long long)R * Cc) return;
  long long r = i / Cc, c = i % Cc;
  out[r * Cc + c] = x[r * Cc + (Cc - 1 - c)];
}

__global__ void k_mean_d(const float* __restrict__ x, float* __restrict__ out,
                         int C, int D, long long S)
{
  long long i = (long long)blockIdx.x * blockDim.x + threadIdx.x;
  if (i >= (long long)C * S) return;
  long long c = i / S, s = i % S;
  float acc = 0.f;
  for (int d = 0; d < D; ++d) acc += x[(c * D + d) * S + s];
  out[i] = acc / (float)D;
}

__global__ void k_dwconv3x3(const float* __restrict__ x, const float* __restrict__ w,
                            const float* __restrict__ b, float* __restrict__ out,
                            int C, int H, int W, int act)
{
  long long i = (long long)blockIdx.x * blockDim.x + threadIdx.x;
  if (i >= (long long)C * H * W) return;
  int wo = (int)(i % W), ho = (int)((i / W) % H), c = (int)(i / ((long long)W * H));
  float acc = b[c];
#pragma unroll
  for (int kh = 0; kh < 3; ++kh)
#pragma unroll
    for (int kw = 0; kw < 3; ++kw) {
      int hh = ho + kh - 1, ww = wo + kw - 1;
      if (hh >= 0 && hh < H && ww >= 0 && ww < W)
        acc += x[((long long)c * H + hh) * W + ww] * w[c * 9 + kh * 3 + kw];
    }
  out[i] = actf(acc, act);
}

__global__ void k_avgpool2(const float* __restrict__ x, float* __restrict__ out,
                           int C, int H, int W)
{
  int Ho = H / 2, Wo = W / 2;
  long long i = (long long)blockIdx.x * blockDim.x + threadIdx.x;
  if (i >= (long long)C * Ho * Wo) return;
  int wo = (int)(i % Wo), ho = (int)((i / Wo) % Ho), c = (int)(i / ((long long)Wo * Ho));
  const float* p = x + ((long long)c * H + 2 * ho) * W + 2 * wo;
  out[i] = 0.25f * (p[0] + p[1] + p[W] + p[W + 1]);
}

__global__ void k_upsample2x(const float* __restrict__ x, float* __restrict__ out,
                             int C, int Hin, int Win)
{
  int Hout = Hin * 2, Wout = Win * 2;
  long long i = (long long)blockIdx.x * blockDim.x + threadIdx.x;
  if (i >= (long long)C * Hout * Wout) return;
  int wo = (int)(i % Wout), ho = (int)((i / Wout) % Hout), c = (int)(i / ((long long)Wout * Hout));
  float fy = (ho + 0.5f) * 0.5f - 0.5f;
  float fx = (wo + 0.5f) * 0.5f - 0.5f;
  int y0 = (int)floorf(fy), x0 = (int)floorf(fx);
  float wy = fy - y0, wx = fx - x0;
  int y1 = y0 + 1, x1 = x0 + 1;
  y0 = max(0, min(Hin - 1, y0)); y1 = max(0, min(Hin - 1, y1));
  x0 = max(0, min(Win - 1, x0)); x1 = max(0, min(Win - 1, x1));
  const float* p = x + (long long)c * Hin * Win;
  float v00 = p[y0 * Win + x0], v01 = p[y0 * Win + x1];
  float v10 = p[y1 * Win + x0], v11 = p[y1 * Win + x1];
  out[i] = (1 - wy) * ((1 - wx) * v00 + wx * v01) + wy * ((1 - wx) * v10 + wx * v11);
}

// fuse 3 evidential heads: raw (52,S); out flat (4,13,S)
__global__ void k_evid_fuse(const float* __restrict__ h0, const float* __restrict__ h1,
                            const float* __restrict__ h2, float* __restrict__ out, long long S)
{
  long long i = (long long)blockIdx.x * blockDim.x + threadIdx.x;
  if (i >= 4 * 13 * S) return;
  int f = (int)(i / (13 * S));
  long long rem = i % (13 * S);
  int c = (int)(rem / S);
  long long s = rem % S;
  const float* hs[3] = {h0, h1, h2};
  float acc = 0.f;
#pragma unroll
  for (int k = 0; k < 3; ++k) {
    float v = hs[k][(long long)(f * 13 + c) * S + s];
    if (f == 1 || f == 3) v = actf(v, ACT_SOFTPLUS);
    else if (f == 2)      v = actf(v, ACT_SOFTPLUS) + 1.f;
    acc += v;
  }
  out[i] = acc * (1.f / 3.f);
}

// ---------------------------------------------------------------------------
// Host-side orchestration
// ---------------------------------------------------------------------------
#define CFG(n) dim3((unsigned)((((long long)(n)) + 255) / 256)), dim3(256), 0, st

struct Ws {
  char* base; size_t off;
  float* alloc(long long elems) {
    float* p = (float*)(base + off);
    off += ((size_t)elems * sizeof(float) + 255) & ~(size_t)255;
    return p;
  }
  _Float16* alloch(long long elems) {
    _Float16* p = (_Float16*)(base + off);
    off += ((size_t)elems * sizeof(_Float16) + 255) & ~(size_t)255;
    return p;
  }
};

static inline void gemm(hipStream_t st, const _Float16* A, const _Float16* B, float* C,
                        const float* bias, int M, int N, int K,
                        int lda, int ldb, int ldc, int tc, int act)
{
  dim3 grid((N + BN - 1) / BN, (M + BM - 1) / BM);
  k_gemm_wmma<<<grid, 256, 0, st>>>(A, B, C, bias, M, N, K, lda, ldb, ldc, tc, act);
}

// convert an f32 tensor (weights / activations) to f16 scratch
static _Float16* cvth(hipStream_t st, Ws& ws, const float* src, long long n)
{
  _Float16* d = ws.alloch(n);
  k_cvt_h<<<CFG(n)>>>(src, d, n);
  return d;
}

// ---- parameter cursor (tree-order traversal of the params pytree) ----
struct PCur { void* const* din; int i; const float* next() { return (const float*)din[i++]; } };
struct ConvP { const float *b, *w; };
struct GnP   { const float *b, *w; };
struct MambaP { const float *A_log, *D, *W_dt, *W_in, *W_out, *W_x, *b_dt, *conv_b, *conv_w; };
struct ResP  { ConvP c1, c2; GnP g1, g2; };
struct C3B   { ConvP c1, c2; GnP g1, g2; };
struct HeadP { ConvP c; GnP g; ConvP head; };
struct MBlk  { ConvP dw; MambaP m_bwd, m_fwd, mt2_bwd, mt2_fwd, mt_bwd, mt_fwd; ConvP pw; ResP res1, res2; };

static ConvP fetchConv(PCur& p) { ConvP c; c.b = p.next(); c.w = p.next(); return c; }
static GnP   fetchGn(PCur& p)   { GnP g; g.b = p.next(); g.w = p.next(); return g; }
static MambaP fetchMamba(PCur& p) {
  MambaP m; m.A_log = p.next(); m.D = p.next(); m.W_dt = p.next(); m.W_in = p.next();
  m.W_out = p.next(); m.W_x = p.next(); m.b_dt = p.next(); m.conv_b = p.next(); m.conv_w = p.next();
  return m;
}
static ResP fetchRes(PCur& p) { ResP r; r.c1 = fetchConv(p); r.c2 = fetchConv(p); r.g1 = fetchGn(p); r.g2 = fetchGn(p); return r; }
static C3B  fetchC3B(PCur& p) { C3B c; c.c1 = fetchConv(p); c.c2 = fetchConv(p); c.g1 = fetchGn(p); c.g2 = fetchGn(p); return c; }
static HeadP fetchHead(PCur& p) { HeadP h; h.c = fetchConv(p); h.g = fetchGn(p); h.head = fetchConv(p); return h; }
static MBlk fetchMBlk(PCur& p) {
  MBlk m; m.dw = fetchConv(p);
  m.m_bwd = fetchMamba(p);  m.m_fwd = fetchMamba(p);
  m.mt2_bwd = fetchMamba(p); m.mt2_fwd = fetchMamba(p);
  m.mt_bwd = fetchMamba(p);  m.mt_fwd = fetchMamba(p);
  m.pw = fetchConv(p); m.res1 = fetchRes(p); m.res2 = fetchRes(p);
  return m;
}

// ---- model helpers (Ws passed by value => scratch auto-reclaimed) ----
static void conv2d3x3(hipStream_t st, const ConvP& p, const float* xin,
                      int Cin, int Cout, int H, int W, float* out, Ws ws)
{
  long long S = (long long)H * W; int Kc = Cin * 9;
  _Float16* pat = ws.alloch(S * Kc);
  _Float16* wh  = cvth(st, ws, p.w, (long long)Cout * Kc);
  k_im2col<<<CFG(S * Kc)>>>(xin, pat, Cin, 1, H, W, 1, 3, 3, S, S);
  gemm(st, pat, wh, out, p.b, (int)S, Cout, Kc, Kc, Kc, (int)S, 1, ACT_NONE);
}

static void conv3d_block(hipStream_t st, const C3B& p, const float* xin,
                         int D, int H, int W, float* out, Ws ws)
{
  const int C = 128;
  long long S = (long long)H * W, M = (long long)D * S;
  int Kc = C * 27;
  _Float16* pat = ws.alloch(M * Kc);
  float* t1  = ws.alloc((long long)C * M);
  float* t2  = ws.alloc((long long)C * M);
  _Float16* w1h = cvth(st, ws, p.c1.w, (long long)C * Kc);
  _Float16* w2h = cvth(st, ws, p.c2.w, (long long)C * Kc);
  k_im2col<<<CFG(M * Kc)>>>(xin, pat, C, D, H, W, 3, 3, 3, M, S);
  gemm(st, pat, w1h, t1, p.c1.b, (int)M, C, Kc, Kc, Kc, (int)M, 1, ACT_NONE);
  k_groupnorm<<<4, 256, 0, st>>>(t1, t2, p.g1.w, p.g1.b, C, M, 4, ACT_RELU);
  k_im2col<<<CFG(M * Kc)>>>(t2, pat, C, D, H, W, 3, 3, 3, M, S);
  gemm(st, pat, w2h, t1, p.c2.b, (int)M, C, Kc, Kc, Kc, (int)M, 1, ACT_NONE);
  k_groupnorm<<<4, 256, 0, st>>>(t1, out, p.g2.w, p.g2.b, C, M, 4, ACT_RELU);
}

static void resconv(hipStream_t st, const ResP& p, const float* xin,
                    int H, int W, float* out, Ws ws)
{
  const int C = 128; long long S = (long long)H * W, n = (long long)C * S;
  float* t1 = ws.alloc(n); float* t2 = ws.alloc(n);
  conv2d3x3(st, p.c1, xin, C, C, H, W, t1, ws);
  k_groupnorm<<<4, 256, 0, st>>>(t1, t2, p.g1.w, p.g1.b, C, S, 4, ACT_GELU);
  conv2d3x3(st, p.c2, t2, C, C, H, W, t1, ws);
  k_groupnorm<<<4, 256, 0, st>>>(t1, t2, p.g2.w, p.g2.b, C, S, 4, ACT_NONE);
  k_add_act<<<CFG(n)>>>(xin, t2, out, ACT_GELU, n);
}

// xh: (L,128) f16 row-major input; out: (128,L) f32 channel-major (transC)
static void mamba_run(hipStream_t st, const MambaP& p, const _Float16* xh,
                      int L, float* out, Ws ws)
{
  long long n1 = (long long)L * 256;
  _Float16* Winh = cvth(st, ws, p.W_in, 512 * 128);
  float* xz = ws.alloc((long long)L * 512);
  gemm(st, xh, Winh, xz, nullptr, L, 512, 128, 128, 128, 512, 0, ACT_NONE);

  float* xc      = ws.alloc(n1);
  _Float16* xch  = ws.alloch(n1);
  k_conv1d_silu<<<CFG(n1)>>>(xz, p.conv_w, p.conv_b, xc, xch, L);

  _Float16* Wxh = cvth(st, ws, p.W_x, 40 * 256);
  float* xdb = ws.alloc((long long)L * 40);
  gemm(st, xch, Wxh, xdb, nullptr, L, 40, 256, 256, 256, 40, 0, ACT_NONE);

  _Float16* xdbh = cvth(st, ws, xdb, (long long)L * 40);
  _Float16* Wdth = cvth(st, ws, p.W_dt, 256 * 8);
  float* dtl = ws.alloc(n1);
  gemm(st, xdbh, Wdth, dtl, p.b_dt, L, 256, 8, 40, 8, 256, 0, ACT_SOFTPLUS);

  float* ys = ws.alloc(n1);
  k_scan<<<1, 256, 0, st>>>(dtl, xdb, xc, p.A_log, p.D, ys, L);

  _Float16* yh = ws.alloch(n1);
  k_gate_h<<<CFG(n1)>>>(ys, xz, yh, L);

  _Float16* Wouth = cvth(st, ws, p.W_out, 128 * 256);
  gemm(st, yh, Wouth, out, nullptr, L, 128, 256, 256, 256, L, 1, ACT_NONE);
}

static void bidir(hipStream_t st, const MambaP& pf, const MambaP& pb,
                  const float* xin, int L, float* out, Ws ws)
{
  const int C = 128; long long n = (long long)C * L;
  _Float16* xh = ws.alloch(n);
  k_cvt_tr_h<<<CFG(n)>>>(xin, xh, C, L);            // (C,L) f32 -> (L,C) f16
  float* f = ws.alloc(n);
  mamba_run(st, pf, xh, L, f, ws);
  _Float16* xhr = ws.alloch(n);
  k_fliprows_h<<<CFG(n)>>>(xh, xhr, L, C);          // reverse sequence
  float* br = ws.alloc(n);
  mamba_run(st, pb, xhr, L, br, ws);
  float* bf = ws.alloc(n);
  k_flipcols<<<CFG(n)>>>(br, bf, C, L);             // un-reverse output (C,L)
  k_axpby<<<CFG(n)>>>(f, bf, out, 0.5f, 0.5f, n);
}

static void mamba_block(hipStream_t st, const MBlk& p, const float* xin,
                        int H, int W, float* out, Ws ws)
{
  const int C = 128; long long S = (long long)H * W, n = (long long)C * S;
  float* xs = ws.alloc(n);
  bidir(st, p.m_fwd, p.m_bwd, xin, (int)S, xs, ws);
  float* xl = ws.alloc(n);
  {
    Ws w2 = ws;
    float* r1 = w2.alloc(n);
    resconv(st, p.res1, xs, H, W, r1, w2);
    float* r2 = w2.alloc(n);
    k_dwconv3x3<<<CFG(n)>>>(r1, p.dw.w, p.dw.b, r2, C, H, W, ACT_GELU);
    float* r3 = w2.alloc(n);
    _Float16* r2h = w2.alloch(n);
    k_cvt_tr_h<<<CFG(n)>>>(r2, r2h, C, S);
    _Float16* pwh = cvth(st, w2, p.pw.w, 128 * 128);
    gemm(st, r2h, pwh, r3, p.pw.b, (int)S, C, C, 128, 128, (int)S, 1, ACT_NONE);
    resconv(st, p.res2, r3, H, W, xl, w2);
  }
  float* xt = ws.alloc(n);
  bidir(st, p.mt_fwd, p.mt_bwd, xin, (int)S, xt, ws);
  int H2 = H / 2, W2 = W / 2; long long S2 = S / 4;
  float* xds = ws.alloc((long long)C * S2);
  k_avgpool2<<<CFG(C * S2)>>>(xin, xds, C, H, W);
  float* xtd = ws.alloc((long long)C * S2);
  bidir(st, p.mt2_fwd, p.mt2_bwd, xds, (int)S2, xtd, ws);
  float* xtu = ws.alloc(n);
  k_upsample2x<<<CFG(n)>>>(xtd, xtu, C, H2, W2);
  k_axpby<<<CFG(n)>>>(xt, xtu, xt, 0.5f, 0.5f, n);
  k_add3<<<CFG(n)>>>(xs, xl, xt, out, n);
}

static void evid_head(hipStream_t st, const HeadP& hp, const float* xin,
                      int H, int W, float* raw, Ws ws)
{
  const int C = 128; long long S = (long long)H * W;
  float* t1 = ws.alloc((long long)C * S);
  float* t2 = ws.alloc((long long)C * S);
  conv2d3x3(st, hp.c, xin, C, C, H, W, t1, ws);
  k_groupnorm<<<4, 256, 0, st>>>(t1, t2, hp.g.w, hp.g.b, C, S, 4, ACT_RELU);
  _Float16* t2h = ws.alloch((long long)C * S);
  k_cvt_tr_h<<<CFG(C * S)>>>(t2, t2h, C, S);
  _Float16* hwh = cvth(st, ws, hp.head.w, 52 * 128);
  gemm(st, t2h, hwh, raw, hp.head.b, (int)S, 52, C, 128, 128, (int)S, 1, ACT_NONE);
}

extern "C" void kernel_launch(void* const* d_in, const int* in_sizes, int n_in,
                              void* d_out, int out_size, void* d_ws, size_t ws_size,
                              hipStream_t stream)
{
  (void)in_sizes; (void)n_in; (void)out_size; (void)ws_size;
  hipStream_t st = stream;
  const float* x = (const float*)d_in[0];    // (1,4,15,64,64)

  PCur pc{d_in, 1};
  // params pytree leaves in sorted-key order:
  HeadP aux0 = fetchHead(pc), aux1 = fetchHead(pc), mainH = fetchHead(pc);  // dec
  C3B  fconv = fetchC3B(pc);  MBlk fmamba = fetchMBlk(pc);                  // enc.final_*
  C3B  l0conv = fetchC3B(pc); MBlk l0m = fetchMBlk(pc);                     // enc.layers[0]
  C3B  l1conv = fetchC3B(pc); MBlk l1m = fetchMBlk(pc);                     // enc.layers[1]
  ConvP inp = fetchConv(pc);                                                 // in_proj
  C3B  pre = fetchC3B(pc);                                                   // pre

  Ws ws{(char*)d_ws, 0};
  const int C = 128;
  const long long S = 4096;     // 64x64
  const long long M0 = 4 * S;   // D=4 depth frames

  float* feat5 = ws.alloc((long long)C * M0);   // (C, D=4, S)
  {
    Ws w2 = ws;
    int K0 = 15 * 27;
    _Float16* pat = w2.alloch(M0 * K0);
    _Float16* wh  = cvth(st, w2, inp.w, (long long)C * K0);
    float* t = w2.alloc((long long)C * M0);
    // input layout (T=4, Cin=15, S): c-stride = S, d-stride = 15*S
    k_im2col<<<CFG(M0 * K0)>>>(x, pat, 15, 4, 64, 64, 3, 3, 3, S, 15 * S);
    gemm(st, pat, wh, t, inp.b, (int)M0, C, K0, K0, K0, (int)M0, 1, ACT_NONE);
    conv3d_block(st, pre, t, 4, 64, 64, feat5, w2);
  }

  float* skip0 = ws.alloc((long long)C * S);
  float* skip1 = ws.alloc((long long)C * S);
  float* feat  = ws.alloc((long long)C * S);

  { // encoder layer 0 (D=4)
    Ws w2 = ws;
    float* t5 = w2.alloc((long long)C * M0);
    conv3d_block(st, l0conv, feat5, 4, 64, 64, t5, w2);
    float* xt = w2.alloc((long long)C * S);
    k_mean_d<<<CFG(C * S)>>>(t5, xt, C, 4, S);
    float* mb = w2.alloc((long long)C * S);
    mamba_block(st, l0m, xt, 64, 64, mb, w2);
    k_axpby<<<CFG(C * S)>>>(mb, xt, skip0, 1.f, 1.f, (long long)C * S);
  }
  { // encoder layer 1 (D=1)
    Ws w2 = ws;
    float* t5 = w2.alloc((long long)C * S);
    conv3d_block(st, l1conv, skip0, 1, 64, 64, t5, w2);
    float* xt = w2.alloc((long long)C * S);
    k_mean_d<<<CFG(C * S)>>>(t5, xt, C, 1, S);
    float* mb = w2.alloc((long long)C * S);
    mamba_block(st, l1m, xt, 64, 64, mb, w2);
    k_axpby<<<CFG(C * S)>>>(mb, xt, skip1, 1.f, 1.f, (long long)C * S);
  }
  { // final conv + mamba
    Ws w2 = ws;
    float* t5 = w2.alloc((long long)C * S);
    conv3d_block(st, fconv, skip1, 1, 64, 64, t5, w2);
    float* xt = w2.alloc((long long)C * S);
    k_mean_d<<<CFG(C * S)>>>(t5, xt, C, 1, S);
    float* mb = w2.alloc((long long)C * S);
    mamba_block(st, fmamba, xt, 64, 64, mb, w2);
    k_axpby<<<CFG(C * S)>>>(mb, xt, feat, 1.f, 1.f, (long long)C * S);
  }

  // evidential heads + fusion -> (4,13,64,64)
  float* raw0 = ws.alloc(52 * S);
  float* raw1 = ws.alloc(52 * S);
  float* raw2 = ws.alloc(52 * S);
  {
    Ws w2 = ws;
    evid_head(st, aux0, skip0, 64, 64, raw0, w2);
    evid_head(st, aux1, skip1, 64, 64, raw1, w2);
    evid_head(st, mainH, feat, 64, 64, raw2, w2);
  }
  k_evid_fuse<<<CFG(4 * 13 * S)>>>(raw0, raw1, raw2, (float*)d_out, S);
}
